// gate_using_local_global_1812476199810
// MI455X (gfx1250) — matrix operand order repaired and checked by reading, not compiled, not run on hardware
//
#include <hip/hip_runtime.h>

typedef __attribute__((ext_vector_type(16))) __bf16 v16bf;
typedef __attribute__((ext_vector_type(8)))  float  v8f;

#define TM 128
#define TN 128
#define TK 32
#define LDH 40   // LDS row stride in ushorts (32 data + 8 pad) -> conflict-free b128 reads

// pack two fp32 -> packed bf16x2 (round-half-up + byte-select). 3 VALU per 2 elements.
__device__ __forceinline__ unsigned int pack_bf16x2(float lo, float hi) {
  unsigned int a = __float_as_uint(lo) + 0x8000u;
  unsigned int b = __float_as_uint(hi) + 0x8000u;
  return __builtin_amdgcn_perm(b, a, 0x07060302u);  // {b[3],b[2],a[3],a[2]} = {bf16(hi),bf16(lo)}
}

__device__ __forceinline__ unsigned short f32_bf16(float f) {
  unsigned int u = __float_as_uint(f) + 0x8000u;
  return (unsigned short)(u >> 16);
}

union FragU { uint4 q[2]; v16bf v; };

// A fragment 16x32 (MxK): lanes 0-15 row M=r hold K {0..7,16..23}; lanes 16-31 hold K {8..15,24..31}
__device__ __forceinline__ v16bf frag_a(const unsigned short* buf, int row0, int lane) {
  const int r = lane & 15, hi = lane >> 4;
  const unsigned short* p = buf + (row0 + r) * LDH + hi * 8;
  FragU u;
  u.q[0] = *reinterpret_cast<const uint4*>(p);
  u.q[1] = *reinterpret_cast<const uint4*>(p + 16);
  return u.v;
}

// B fragment 32x16 (KxN): lanes 0-15 col N=c hold K {0..7,16..23}; lanes 16-31 hold K {8..15,24..31}
__device__ __forceinline__ v16bf frag_b(const unsigned short* buf, int col0, int lane) {
  const int c = lane & 15, hi = lane >> 4;
  const unsigned short* p = buf + (col0 + c) * LDH + hi * 8;
  FragU u;
  u.q[0] = *reinterpret_cast<const uint4*>(p);
  u.q[1] = *reinterpret_cast<const uint4*>(p + 16);
  return u.v;
}

// Generic C[M x N] = alpha * A[M x K] * (transB ? Bsrc[N x K]^T : Bsrc[K x N])
// ABF/BBF: operand already bf16 (ushort) in memory -> staging is a raw copy (no VALU convert).
template <bool ABF, bool BBF>
__global__ __launch_bounds__(256)
void gemm_wmma(const void* __restrict__ Av, long long aBS, int lda,
               const void* __restrict__ Bv, long long bBS, int ldb,
               float* __restrict__ C, long long cBS, int ldc,
               int K, int transB, float alpha)
{
  __shared__ __align__(16) unsigned short Ash[2][TM * LDH];
  __shared__ __align__(16) unsigned short Bsh[2][TN * LDH];

  const int tid  = threadIdx.x;
  const int lane = tid & 31;
  const int wave = tid >> 5;
  const int wm   = wave >> 1;   // 0..3 : 32-row strip
  const int wn   = wave & 1;    // 0..1 : 64-col strip

  const int m0 = blockIdx.y * TM;
  const int n0 = blockIdx.x * TN;

  const float*          Af = ABF ? nullptr : ((const float*)Av + (long long)blockIdx.z * aBS);
  const unsigned short* Ah = ABF ? ((const unsigned short*)Av + (long long)blockIdx.z * aBS) : nullptr;
  const float*          Bf = BBF ? nullptr : ((const float*)Bv + (long long)blockIdx.z * bBS);
  const unsigned short* Bh = BBF ? ((const unsigned short*)Bv + (long long)blockIdx.z * bBS) : nullptr;
  C += (long long)blockIdx.z * cBS;

  v8f acc[2][4];
  #pragma unroll
  for (int i = 0; i < 2; ++i)
    #pragma unroll
    for (int j = 0; j < 4; ++j)
      #pragma unroll
      for (int e = 0; e < 8; ++e) acc[i][j][e] = 0.0f;

  float4 aR[4];  uint2 aU[4];             // one set live per instantiation
  float4 bR4[4]; uint2 bU[4];
  float  bRt[16]; unsigned short bRh[16];

  auto fetchA = [&](int kt) {
    const int k0 = kt * TK;
    #pragma unroll
    for (int i = 0; i < 4; ++i) {
      const int f = tid + 256 * i;         // 1024 groups of 4 elements per 128x32 tile
      const int row = f >> 3;
      const int col = (f & 7) << 2;
      if constexpr (ABF)
        aU[i] = *reinterpret_cast<const uint2*>(Ah + (long long)(m0 + row) * lda + (k0 + col));
      else
        aR[i] = *reinterpret_cast<const float4*>(Af + (long long)(m0 + row) * lda + (k0 + col));
    }
  };
  auto commitA = [&](unsigned short* buf) {
    #pragma unroll
    for (int i = 0; i < 4; ++i) {
      const int f = tid + 256 * i;
      const int row = f >> 3;
      const int col = (f & 7) << 2;
      uint2 u;
      if constexpr (ABF) u = aU[i];
      else { u.x = pack_bf16x2(aR[i].x, aR[i].y); u.y = pack_bf16x2(aR[i].z, aR[i].w); }
      *reinterpret_cast<uint2*>(buf + row * LDH + col) = u;
    }
  };
  auto fetchB = [&](int kt) {
    const int k0 = kt * TK;
    if (transB) {  // Bsrc is [N x K] row-major: stage rows directly (same pattern as A)
      #pragma unroll
      for (int i = 0; i < 4; ++i) {
        const int f = tid + 256 * i;
        const int row = f >> 3;              // n
        const int col = (f & 7) << 2;        // k
        if constexpr (BBF)
          bU[i] = *reinterpret_cast<const uint2*>(Bh + (long long)(n0 + row) * ldb + (k0 + col));
        else
          bR4[i] = *reinterpret_cast<const float4*>(Bf + (long long)(n0 + row) * ldb + (k0 + col));
      }
    } else {       // Bsrc is [K x N]: transpose into LDS (coalesced reads across n)
      const int n  = tid & 127;
      const int kk = (tid >> 7) * 16;
      #pragma unroll
      for (int j = 0; j < 16; ++j) {
        if constexpr (BBF) bRh[j] = Bh[(long long)(k0 + kk + j) * ldb + (n0 + n)];
        else               bRt[j] = Bf[(long long)(k0 + kk + j) * ldb + (n0 + n)];
      }
    }
  };
  auto commitB = [&](unsigned short* buf) {
    if (transB) {
      #pragma unroll
      for (int i = 0; i < 4; ++i) {
        const int f = tid + 256 * i;
        const int row = f >> 3;
        const int col = (f & 7) << 2;
        uint2 u;
        if constexpr (BBF) u = bU[i];
        else { u.x = pack_bf16x2(bR4[i].x, bR4[i].y); u.y = pack_bf16x2(bR4[i].z, bR4[i].w); }
        *reinterpret_cast<uint2*>(buf + row * LDH + col) = u;
      }
    } else {
      const int n  = tid & 127;
      const int kk = (tid >> 7) * 16;
      #pragma unroll
      for (int j = 0; j < 16; j += 2) {
        unsigned int u;
        if constexpr (BBF) u = (unsigned int)bRh[j] | ((unsigned int)bRh[j + 1] << 16);
        else               u = pack_bf16x2(bRt[j], bRt[j + 1]);
        *reinterpret_cast<unsigned int*>(buf + n * LDH + kk + j) = u;
      }
    }
  };

  const int KT = K / TK;
  fetchA(0); fetchB(0);
  commitA(Ash[0]); commitB(Bsh[0]);
  __syncthreads();

  for (int kt = 0; kt < KT; ++kt) {
    const int cur = kt & 1;
    const bool more = (kt + 1) < KT;
    if (more) { fetchA(kt + 1); fetchB(kt + 1); }   // global loads overlap WMMA below

    v16bf af[2], bfr[4];
    #pragma unroll
    for (int i = 0; i < 2; ++i) af[i] = frag_a(Ash[cur], wm * 32 + i * 16, lane);
    #pragma unroll
    for (int j = 0; j < 4; ++j) bfr[j] = frag_b(Bsh[cur], wn * 64 + j * 16, lane);

    #pragma unroll
    for (int i = 0; i < 2; ++i)
      #pragma unroll
      for (int j = 0; j < 4; ++j)
        acc[i][j] = __builtin_amdgcn_wmma_f32_16x16x32_bf16(
            false, af[i], false, bfr[j], (short)0, acc[i][j], false, false);

    if (more) { commitA(Ash[cur ^ 1]); commitB(Bsh[cur ^ 1]); }  // write other buffer
    __syncthreads();
  }

  // C layout: lanes 0-15 -> (M=e, N=lane); lanes 16-31 -> (M=e+8, N=lane-16)
  const int r = lane & 15, hi = lane >> 4;
  #pragma unroll
  for (int i = 0; i < 2; ++i) {
    #pragma unroll
    for (int j = 0; j < 4; ++j) {
      const int rb = m0 + wm * 32 + i * 16 + hi * 8;
      const int cb = n0 + wn * 64 + j * 16 + r;
      #pragma unroll
      for (int e = 0; e < 8; ++e)
        C[(long long)(rb + e) * ldc + cb] = acc[i][j][e] * alpha;
    }
  }
}

// fp32 -> bf16 one-shot conversion, 8 elements/thread, fully coalesced b128 in / b128 out
extern "C" __global__ __launch_bounds__(256)
void convert_f32_bf16(const float* __restrict__ src, unsigned short* __restrict__ dst) {
  const long long i = ((long long)blockIdx.x * 256 + threadIdx.x) * 8;
  float4 f0 = *reinterpret_cast<const float4*>(src + i);
  float4 f1 = *reinterpret_cast<const float4*>(src + i + 4);
  uint4 o;
  o.x = pack_bf16x2(f0.x, f0.y);
  o.y = pack_bf16x2(f0.z, f0.w);
  o.z = pack_bf16x2(f1.x, f1.y);
  o.w = pack_bf16x2(f1.z, f1.w);
  *reinterpret_cast<uint4*>(dst + i) = o;
}

// BD[b][i][j] = (i/64 == j/64) ? KV[b][j][i] : 0   (blockdiag of kv_h^T)
extern "C" __global__ void build_bd_kernel(const float* __restrict__ KV, float* __restrict__ BD) {
  const long long idx = (long long)blockIdx.x * blockDim.x + threadIdx.x;  // 8*512*512 total
  const int b   = (int)(idx >> 18);
  const int rem = (int)(idx & 262143);
  const int i = rem >> 9, j = rem & 511;
  float v = 0.0f;
  if ((i >> 6) == (j >> 6)) v = KV[((long long)b << 18) + (long long)j * 512 + i];
  BD[idx] = v;
}

extern "C" void kernel_launch(void* const* d_in, const int* in_sizes, int n_in,
                              void* d_out, int out_size, void* d_ws, size_t ws_size,
                              hipStream_t stream) {
  const float* xl = (const float*)d_in[0];   // [8,512,8192]
  const float* xg = (const float*)d_in[1];   // [8,512,8192]
  const float* Wq = (const float*)d_in[2];   // [512,512]
  const float* Wk = (const float*)d_in[3];
  const float* Wv = (const float*)d_in[4];
  const float* Wo = (const float*)d_in[5];
  float* out = (float*)d_out;                // [8,512,8192]

  const long long P2 = 512ll * 512;          // 262144
  const long long XB = 512ll * 8192;         // per-batch x stride (elements)

  // workspace: bf16 copies of both activations (67 MB each) + 3 reusable fp32 [8,512,512]
  unsigned short* xgh = (unsigned short*)d_ws;
  unsigned short* xlh = xgh + 8 * XB;
  float* buf0 = (float*)(xlh + 8 * XB);
  float* buf1 = buf0 + 8 * P2;
  float* buf2 = buf1 + 8 * P2;

  float* G    = buf0;
  float* T1t  = buf1;
  float* KVm  = buf2;
  float* BD   = buf0;   // G dead after step 2
  float* S2   = buf1;   // T1t dead after step 3
  float* Weff = buf2;   // KV dead after step 4

  dim3 blk(256);
  // 0) one-shot fp32 -> bf16 of both activation tensors (8*512*8192 = 16384*256*8 elements)
  convert_f32_bf16<<<dim3(16384), blk, 0, stream>>>(xg, xgh);
  convert_f32_bf16<<<dim3(16384), blk, 0, stream>>>(xl, xlh);
  // 1) G[b] = Xg[b] @ Xg[b]^T            (M=N=512, K=8192, both bf16 -> copy staging)
  gemm_wmma<true, true><<<dim3(4, 4, 8), blk, 0, stream>>>(xgh, XB, 8192, xgh, XB, 8192, G, P2, 512, 8192, 1, 1.0f);
  // 2) T1t[b] = Wv @ G[b]                (G symmetric => T1t = (G@Wv^T)^T)
  gemm_wmma<false, false><<<dim3(4, 4, 8), blk, 0, stream>>>(Wv, 0, 512, G, P2, 512, T1t, P2, 512, 512, 0, 1.0f);
  // 3) KV[b] = Wk @ T1t[b]^T = Wk G Wv^T (diag 64x64 blocks are kv_h)
  gemm_wmma<false, false><<<dim3(4, 4, 8), blk, 0, stream>>>(Wk, 0, 512, T1t, P2, 512, KVm, P2, 512, 512, 1, 1.0f);
  // 4) BD[b] = blockdiag(kv_h^T)
  build_bd_kernel<<<dim3(8192), blk, 0, stream>>>(KVm, BD);
  // 5) S2[b] = BD[b] @ Wq
  gemm_wmma<false, false><<<dim3(4, 4, 8), blk, 0, stream>>>(BD, P2, 512, Wq, 0, 512, S2, P2, 512, 512, 0, 1.0f);
  // 6) Weff[b] = Wo @ S2[b] / 64
  gemm_wmma<false, false><<<dim3(4, 4, 8), blk, 0, stream>>>(Wo, 0, 512, S2, P2, 512, Weff, P2, 512, 512, 0, 1.0f / 64.0f);
  // 7) out[b] = Weff[b] @ Xl[b]          (M=512, N=8192, K=512, B is bf16)
  gemm_wmma<false, true><<<dim3(64, 4, 8), blk, 0, stream>>>(Weff, P2, 512, xlh, XB, 8192, out, XB, 8192, 512, 0, 1.0f);
}
